// SimpleGATGVPCrossModel_88235808129711
// MI455X (gfx1250) — matrix operand order repaired and checked by reading, not compiled, not run on hardware
//
#include <hip/hip_runtime.h>
#include <hip/hip_bf16.h>

// ---------------------------------------------------------------------------
// CDNA5 / gfx1250 implementation of SimpleGATGVPCrossModel forward.
// All dense GEMMs and both cross-attentions use v_wmma_f32_16x16x32_f16
// (f16 in / f32 accumulate).  Ragged edge ops use VALU + f32 atomics.
// Workspace: bump-allocated from d_ws (~230 MB needed).
// ---------------------------------------------------------------------------

typedef _Float16 v16h __attribute__((ext_vector_type(16)));
typedef _Float16 h8   __attribute__((ext_vector_type(8)));
typedef float    v8f  __attribute__((ext_vector_type(8)));

#define SHUF16(a,b) __builtin_shufflevector((a),(b),0,1,2,3,4,5,6,7,8,9,10,11,12,13,14,15)

static __device__ __forceinline__ float lrelu(float x){ return x > 0.f ? x : 0.2f*x; }
static __device__ __forceinline__ v8f vzero8(){ v8f z = {0.f,0.f,0.f,0.f,0.f,0.f,0.f,0.f}; return z; }

static __device__ __forceinline__ void atomicMaxF(float* a, float v){
  unsigned int* u = (unsigned int*)a;
  unsigned int old = *u;
  while (__uint_as_float(old) < v) {
    unsigned int assumed = old;
    old = atomicCAS(u, assumed, __float_as_uint(v));
    if (old == assumed) break;
  }
}

// ---------------------------------------------------------------------------
// Conversion / fill kernels
// ---------------------------------------------------------------------------
__global__ void k_fillf(float* p, float v, int n){
  int i = blockIdx.x*blockDim.x + threadIdx.x;
  if (i < n) p[i] = v;
}
__global__ void k_filli(int* p, int v, int n){
  int i = blockIdx.x*blockDim.x + threadIdx.x;
  if (i < n) p[i] = v;
}
// f32 (rows x cols) -> f16 (rows x Kp) zero-padded
__global__ void k_cvtA(const float* __restrict__ S, int cols, _Float16* __restrict__ D,
                       int Kp, int total){
  int i = blockIdx.x*blockDim.x + threadIdx.x;
  if (i >= total) return;
  int r = i / Kp, k = i % Kp;
  D[i] = (k < cols) ? (_Float16)S[(size_t)r*cols + k] : (_Float16)0.f;
}
// weight W (K x N) f32 -> Bt (N x Kp) f16 zero-padded (transposed for WMMA B)
__global__ void k_cvtT(const float* __restrict__ W, int K, int N,
                       _Float16* __restrict__ D, int Kp){
  int i = blockIdx.x*blockDim.x + threadIdx.x;
  int total = N * Kp;
  if (i >= total) return;
  int n = i / Kp, k = i % Kp;
  D[i] = (k < K) ? (_Float16)W[(size_t)k*N + n] : (_Float16)0.f;
}
// f32 (rows x cols) -> f16 slice of (rows x dld) at column dcol (for concat)
__global__ void k_cvtOff(const float* __restrict__ S, int cols,
                         _Float16* __restrict__ D, int dld, int dcol, int total){
  int i = blockIdx.x*blockDim.x + threadIdx.x;
  if (i >= total) return;
  int r = i / cols, c = i % cols;
  D[(size_t)r*dld + dcol + c] = (_Float16)S[i];
}
// column means of (rows x cols) f32
__global__ void k_colmean(const float* __restrict__ S, int rows, int cols, float* out){
  __shared__ float red[256];
  int f = blockIdx.x, t = threadIdx.x;
  float a = 0.f;
  for (int r = t; r < rows; r += 256) a += S[(size_t)r*cols + f];
  red[t] = a; __syncthreads();
  for (int o = 128; o > 0; o >>= 1){ if (t < o) red[t] += red[t+o]; __syncthreads(); }
  if (t == 0) out[f] = red[0] / (float)rows;
}

// ---------------------------------------------------------------------------
// WMMA GEMM:  C[M,N] = A[M,Kp](f16) @ Bt[N,Kp]^T(f16)  (+bias, +lrelu)
// 4 waves per block; each wave computes a 32 x (TN*16) tile (2 M-fragments x
// TN N-fragments, all unconditional -> accumulators stay pinned).
// M multiple of 32, N multiple of TN*16, Kp multiple of 32.
// ---------------------------------------------------------------------------
template <int TN>
__global__ __launch_bounds__(128)
void k_gemm(const _Float16* __restrict__ A, const _Float16* __restrict__ Bt,
            const float* __restrict__ bias, float* __restrict__ C,
            int M, int N, int Kp, int act){
  int ngrp = (N >> 4) / TN;
  int mtiles = M >> 5;
  int mt = (blockIdx.x / ngrp) * 4 + (threadIdx.x >> 5);
  int ng = blockIdx.x % ngrp;
  if (mt >= mtiles) return;                 // wave-uniform; no barriers below
  int lane = threadIdx.x & 31;
  int lo = lane & 15, hi = lane >> 4;
  v8f acc0[TN], acc1[TN];
  #pragma unroll
  for (int t = 0; t < TN; ++t){ acc0[t] = vzero8(); acc1[t] = vzero8(); }
  const _Float16* arow0 = A + (size_t)(mt*32 + lo) * Kp;
  const _Float16* arow1 = A + (size_t)(mt*32 + 16 + lo) * Kp;
  for (int k0 = 0; k0 < Kp; k0 += 32){
    const _Float16* ap0 = arow0 + k0 + hi*8;
    h8 x0 = *(const h8*)ap0;
    h8 x1 = *(const h8*)(ap0 + 16);
    v16h af0 = SHUF16(x0, x1);
    const _Float16* ap1 = arow1 + k0 + hi*8;
    h8 y0 = *(const h8*)ap1;
    h8 y1 = *(const h8*)(ap1 + 16);
    v16h af1 = SHUF16(y0, y1);
    if (k0 + 32 < Kp){                      // prefetch next A K-slice (global_prefetch_b8)
      __builtin_prefetch(arow0 + k0 + 32, 0, 3);
      __builtin_prefetch(arow1 + k0 + 32, 0, 3);
    }
    #pragma unroll
    for (int t = 0; t < TN; ++t){
      const _Float16* bp = Bt + (size_t)((ng*TN + t)*16 + lo) * Kp + k0 + hi*8;
      h8 b0 = *(const h8*)bp;
      h8 b1 = *(const h8*)(bp + 16);
      v16h bf = SHUF16(b0, b1);
      acc0[t] = __builtin_amdgcn_wmma_f32_16x16x32_f16(false, af0, false, bf,
                                                       (short)0, acc0[t], false, false);
      acc1[t] = __builtin_amdgcn_wmma_f32_16x16x32_f16(false, af1, false, bf,
                                                       (short)0, acc1[t], false, false);
    }
  }
  #pragma unroll
  for (int t = 0; t < TN; ++t){
    int col = (ng*TN + t)*16 + lo;
    float bv = bias ? bias[col] : 0.f;
    #pragma unroll
    for (int v = 0; v < 8; ++v){
      int r0 = mt*32 + v + 8*hi;
      float z0 = acc0[t][v] + bv;
      if (act) z0 = lrelu(z0);
      C[(size_t)r0*N + col] = z0;
      int r1 = r0 + 16;
      float z1 = acc1[t][v] + bv;
      if (act) z1 = lrelu(z1);
      C[(size_t)r1*N + col] = z1;
    }
  }
}

// ---------------------------------------------------------------------------
// GATv2 edge kernels (edge e >= E is the self-loop of node e-E with mean attr)
// ---------------------------------------------------------------------------
__global__ void k_gat_logit(const float* __restrict__ XL, const float* __restrict__ XR,
                            const int* __restrict__ ei, int E, int Nn,
                            const float* __restrict__ eattr, int edim,
                            const float* __restrict__ emean,
                            const float* __restrict__ We,   // edim x 128
                            const float* __restrict__ att,  // 4 x 32
                            float* __restrict__ LG, float* __restrict__ SM){
  int t = blockIdx.x*blockDim.x + threadIdx.x;
  int Etot = E + Nn;
  if (t >= Etot*4) return;
  int e = t >> 2, h = t & 3;
  int s, d; const float* ea;
  if (e < E){ s = ei[e]; d = ei[E + e]; ea = eattr + (size_t)e*edim; }
  else { s = d = e - E; ea = emean; }
  float acc = 0.f;
  for (int c = 0; c < 32; ++c){
    int col = h*32 + c;
    float ee = 0.f;
    for (int q = 0; q < edim; ++q) ee += ea[q] * We[q*128 + col];
    float v = XL[(size_t)s*128 + col] + XR[(size_t)d*128 + col] + ee;
    acc += lrelu(v) * att[h*32 + c];
  }
  LG[t] = acc;
  atomicMaxF(&SM[d*4 + h], acc);
}
__global__ void k_gat_exp(const float* __restrict__ LG, const int* __restrict__ ei,
                          int E, int Nn, const float* __restrict__ SM,
                          float* __restrict__ EXB, float* __restrict__ SD){
  int t = blockIdx.x*blockDim.x + threadIdx.x;
  int Etot = E + Nn;
  if (t >= Etot*4) return;
  int e = t >> 2, h = t & 3;
  int d = (e < E) ? ei[E + e] : e - E;
  float ex = __expf(LG[t] - SM[d*4 + h]);
  EXB[t] = ex;
  atomicAdd(&SD[d*4 + h], ex);
}
__global__ void k_gat_agg(const float* __restrict__ EXB, const float* __restrict__ SD,
                          const float* __restrict__ XL, const int* __restrict__ ei,
                          int E, int Nn, float* __restrict__ AGG){
  int t = blockIdx.x*blockDim.x + threadIdx.x;
  int Etot = E + Nn;
  if (t >= Etot*128) return;
  int e = t >> 7, c = t & 127, h = c >> 5;
  int s, d;
  if (e < E){ s = ei[e]; d = ei[E + e]; } else { s = d = e - E; }
  float alpha = EXB[e*4 + h] / fmaxf(SD[d*4 + h], 1e-16f);
  atomicAdd(&AGG[(size_t)d*128 + c], alpha * XL[(size_t)s*128 + c]);
}
__global__ void k_add3(const float* __restrict__ AGG, const float* __restrict__ bias,
                       const float* __restrict__ XRES, float* __restrict__ X, int total){
  int i = blockIdx.x*blockDim.x + threadIdx.x;
  if (i < total) X[i] = AGG[i] + bias[i & 127] + XRES[i];
}

// ---------------------------------------------------------------------------
// BatchNorm (training-mode, biased variance)
// ---------------------------------------------------------------------------
__global__ void k_bn_stats(const float* __restrict__ X, int n, int F,
                           float* __restrict__ mu, float* __restrict__ var){
  __shared__ float s1[256], s2[256];
  int f = blockIdx.x, t = threadIdx.x;
  float a = 0.f, b = 0.f;
  for (int r = t; r < n; r += 256){
    float x = X[(size_t)r*F + f]; a += x; b += x*x;
  }
  s1[t] = a; s2[t] = b; __syncthreads();
  for (int o = 128; o > 0; o >>= 1){
    if (t < o){ s1[t] += s1[t+o]; s2[t] += s2[t+o]; } __syncthreads();
  }
  if (t == 0){ float m = s1[0]/n; mu[f] = m; var[f] = s2[0]/n - m*m; }
}
__global__ void k_bn_apply(float* __restrict__ X, int F, const float* __restrict__ mu,
                           const float* __restrict__ var, const float* __restrict__ g,
                           const float* __restrict__ b, int act, int total){
  int i = blockIdx.x*blockDim.x + threadIdx.x;
  if (i >= total) return;
  int c = i % F;
  float y = (X[i] - mu[c]) * rsqrtf(var[c] + 1e-5f) * g[c] + b[c];
  X[i] = act ? lrelu(y) : y;
}

// ---------------------------------------------------------------------------
// Fused masked cross-attention, flash-style, WMMA for QK^T and P.V.
// One wave (block of 32) per 16-query tile (single-wave workgroup: the
// __syncthreads below legally degenerates to S_NOP; trip counts differ per
// wave so multi-wave blocks would have divergent barriers).
// Q16: Nq x 128 f16; K16: Nk x 128 f16; Vt16: 128 x NkP f16 (V transposed,
// zero-padded to NkP = Nk+32).  O: Nq x 128 f32 (unprojected attn output).
// ---------------------------------------------------------------------------
__global__ __launch_bounds__(32)
void k_cross_attn(const _Float16* __restrict__ Q16, const _Float16* __restrict__ K16,
                  const _Float16* __restrict__ Vt16, const int* __restrict__ qb,
                  const int* __restrict__ kb, const int* __restrict__ kstart,
                  const int* __restrict__ kcnt, int Nq, int Nk, int NkP,
                  float* __restrict__ O){
  __shared__ _Float16 pbuf[16][32];
  int lane = threadIdx.x;
  int lo = lane & 15, hi = lane >> 4;
  int qt = blockIdx.x, q0 = qt * 16;
  const float scale = 0.08838834764831845f;   // 1/sqrt(128)

  v16h qf[4];
  {
    const _Float16* qp = Q16 + (size_t)(q0 + lo) * 128;
    #pragma unroll
    for (int kk = 0; kk < 4; ++kk){
      const _Float16* p = qp + kk*32 + hi*8;
      h8 a0 = *(const h8*)p;
      h8 a1 = *(const h8*)(p + 16);
      qf[kk] = SHUF16(a0, a1);
    }
  }
  int   rb[8];
  float rowM[8], rowD[8];
  v8f   oacc[8];
  #pragma unroll
  for (int v = 0; v < 8; ++v){
    rb[v] = qb[q0 + v + 8*hi];
    rowM[v] = -3.0e38f; rowD[v] = 0.f;
  }
  #pragma unroll
  for (int t = 0; t < 8; ++t) oacc[t] = vzero8();

  int b0 = qb[q0], b1 = qb[q0 + 15];
  int kend = 0;
  for (int b = b0; b <= b1; ++b){
    int e = kstart[b] + kcnt[b];
    if (e > kend) kend = e;
  }
  int kbeg = kstart[b0] & ~31;   // 32-aligned so all f16 vector loads stay aligned

  for (int n0 = kbeg; n0 < kend; n0 += 32){
    // ---- scores: S = Q @ K^T for 32 keys (two 16x16 fragments)
    v8f s0 = vzero8(), s1 = vzero8();
    int k0idx = n0 + lo;       if (k0idx >= Nk) k0idx = Nk - 1;
    int k1idx = n0 + 16 + lo;  if (k1idx >= Nk) k1idx = Nk - 1;
    #pragma unroll
    for (int kk = 0; kk < 4; ++kk){
      const _Float16* pk0 = K16 + (size_t)k0idx*128 + kk*32 + hi*8;
      h8 c0 = *(const h8*)pk0; h8 c1 = *(const h8*)(pk0 + 16);
      v16h bf0 = SHUF16(c0, c1);
      const _Float16* pk1 = K16 + (size_t)k1idx*128 + kk*32 + hi*8;
      h8 d0 = *(const h8*)pk1; h8 d1 = *(const h8*)(pk1 + 16);
      v16h bf1 = SHUF16(d0, d1);
      s0 = __builtin_amdgcn_wmma_f32_16x16x32_f16(false, qf[kk], false, bf0, (short)0, s0, false, false);
      s1 = __builtin_amdgcn_wmma_f32_16x16x32_f16(false, qf[kk], false, bf1, (short)0, s1, false, false);
    }
    // ---- mask + online softmax (per-row butterfly over the 16-lane half)
    int key0 = n0 + lo, key1 = n0 + 16 + lo;
    int kb0 = (key0 < kend) ? kb[key0] : -1;
    int kb1 = (key1 < kend) ? kb[key1] : -1;
    float p0[8], p1[8];
    #pragma unroll
    for (int v = 0; v < 8; ++v){
      float x0 = (kb0 == rb[v]) ? s0[v]*scale : -3.0e38f;
      float x1 = (kb1 == rb[v]) ? s1[v]*scale : -3.0e38f;
      float mx = fmaxf(x0, x1);
      mx = fmaxf(mx, __shfl_xor(mx, 1, 16));
      mx = fmaxf(mx, __shfl_xor(mx, 2, 16));
      mx = fmaxf(mx, __shfl_xor(mx, 4, 16));
      mx = fmaxf(mx, __shfl_xor(mx, 8, 16));
      float m2 = fmaxf(rowM[v], mx);
      bool any = m2 > -1.0e30f;
      float sc = any ? __expf(rowM[v] - m2) : 1.f;
      rowD[v] *= sc;
      #pragma unroll
      for (int t = 0; t < 8; ++t) oacc[t][v] *= sc;
      rowM[v] = m2;
      float e0 = any ? __expf(x0 - m2) : 0.f;
      float e1 = any ? __expf(x1 - m2) : 0.f;
      float rs = e0 + e1;
      rs += __shfl_xor(rs, 1, 16);
      rs += __shfl_xor(rs, 2, 16);
      rs += __shfl_xor(rs, 4, 16);
      rs += __shfl_xor(rs, 8, 16);
      rowD[v] += rs;
      p0[v] = e0; p1[v] = e1;
    }
    // ---- C-layout -> A-layout re-distribution of P through LDS
    __syncthreads();
    #pragma unroll
    for (int v = 0; v < 8; ++v){
      pbuf[v + 8*hi][lo]      = (_Float16)p0[v];
      pbuf[v + 8*hi][16 + lo] = (_Float16)p1[v];
    }
    __syncthreads();
    v16h pf;
    {
      const _Float16* pr = &pbuf[lo][hi*8];
      h8 a0 = *(const h8*)pr;
      h8 a1 = *(const h8*)(pr + 16);
      pf = SHUF16(a0, a1);
    }
    // ---- O += P @ V (8 column tiles of 16)
    #pragma unroll
    for (int t = 0; t < 8; ++t){
      const _Float16* vp = Vt16 + (size_t)(t*16 + lo) * NkP + n0 + hi*8;
      h8 b0v = *(const h8*)vp;
      h8 b1v = *(const h8*)(vp + 16);
      v16h vf = SHUF16(b0v, b1v);
      oacc[t] = __builtin_amdgcn_wmma_f32_16x16x32_f16(false, pf, false, vf, (short)0, oacc[t], false, false);
    }
  }
  #pragma unroll
  for (int t = 0; t < 8; ++t){
    #pragma unroll
    for (int v = 0; v < 8; ++v){
      O[(size_t)(q0 + v + 8*hi)*128 + t*16 + lo] = oacc[t][v] / fmaxf(rowD[v], 1e-16f);
    }
  }
}

// ---------------------------------------------------------------------------
// Misc small kernels
// ---------------------------------------------------------------------------
__global__ void k_bseg(const int* __restrict__ batch, int n, int* __restrict__ start,
                       int* __restrict__ cnt){
  int i = blockIdx.x*blockDim.x + threadIdx.x;
  if (i >= n) return;
  int b = batch[i];
  atomicAdd(&cnt[b], 1);
  if (i == 0 || batch[i-1] != b) start[b] = i;
}
__global__ void k_fwmix(float* __restrict__ X, const float* __restrict__ A,
                        const float* __restrict__ FW, int total){
  int i = blockIdx.x*blockDim.x + threadIdx.x;
  if (i >= total) return;
  int d = i & 127;
  float w0 = FW[d*2], w1 = FW[d*2 + 1];
  float m = fmaxf(w0, w1);
  float e0 = __expf(w0 - m), e1 = __expf(w1 - m);
  X[i] = (X[i]*e0 + A[i]*e1) / (e0 + e1);
}
__global__ void k_dotw(const float* __restrict__ X, int M, int C,
                       const float* __restrict__ W, const float* __restrict__ b,
                       float* __restrict__ out){
  int r = blockIdx.x*blockDim.x + threadIdx.x;
  if (r >= M) return;
  float acc = b[0];
  for (int c = 0; c < C; ++c) acc += X[(size_t)r*C + c] * W[c];
  out[r] = acc;
}
__global__ void k_pool_max(const float* __restrict__ g, const int* __restrict__ batch,
                           int n, float* __restrict__ m){
  int i = blockIdx.x*blockDim.x + threadIdx.x;
  if (i < n) atomicMaxF(&m[batch[i]], g[i]);
}
__global__ void k_pool_exp(const float* __restrict__ g, const int* __restrict__ batch,
                           int n, const float* __restrict__ m, float* __restrict__ ex,
                           float* __restrict__ den){
  int i = blockIdx.x*blockDim.x + threadIdx.x;
  if (i >= n) return;
  int b = batch[i];
  float e = __expf(g[i] - m[b]);
  ex[i] = e;
  atomicAdd(&den[b], e);
}
__global__ void k_pool_agg(const float* __restrict__ ex, const float* __restrict__ den,
                           const int* __restrict__ batch, const float* __restrict__ X,
                           int n, float* __restrict__ out){
  int t = blockIdx.x*blockDim.x + threadIdx.x;
  if (t >= n*128) return;
  int i = t >> 7;
  int b = batch[i];
  atomicAdd(&out[(size_t)b*128 + (t & 127)], ex[i] / fmaxf(den[b], 1e-16f) * X[t]);
}

// ---------------------------------------------------------------------------
// Host orchestration
// ---------------------------------------------------------------------------
extern "C" void kernel_launch(void* const* d_in, const int* in_sizes, int n_in,
                              void* d_out, int out_size, void* d_ws, size_t ws_size,
                              hipStream_t stream){
  (void)in_sizes; (void)n_in; (void)out_size; (void)ws_size;
  const int NP = 32768, ND = 2560, EP = 524288, ED = 10240, B = 64;

  // ---- input index map (params flattened in setup_inputs insertion order) ----
  // 0..21   drug_enc: L0{Wl,bl,Wr,br,att,We,bias,Wres} L1{..} bn0{g,b} bn1{g,b} final{W,b}
  // 22..63  prot_gat: L0..L3 (8 each), bn0..bn3 (2 each), final{W,b}
  // 64..77  fusion: m1v,m1o,m2v,m2o,l1 (W,b each), bn{g,b}, l2{W,b}
  // 78..85  ca_d2p q,k,v,o ; 86..93 ca_p2d ; 94 fw_p ; 95 fw_d
  // 96..99  pool_p l1{W,b} l2{W,b} ; 100..103 pool_d
  // 104..117 head l1{W,b} bn1 l2{W,b} bn2 l3{W,b} bn3 l4{W,b}
  // 118 drug_x  119 drug_edge_attr 120 prot_x 121 prot_edge_attr 122 x_p_gvp
  // 123 drug_edge_index 124 prot_edge_index 125 drug_batch 126 prot_batch
  #define PF(i) ((const float*)d_in[(i)])
  #define PI(i) ((const int*)d_in[(i)])
  const int* dEI = PI(123); const int* pEI = PI(124);
  const int* DB = PI(125);  const int* PB = PI(126);

  // ---- workspace bump allocator ----
  char* wp = (char*)d_ws;
  auto alloc = [&](size_t n){ void* r = (void*)wp; wp += (n + 255) & ~(size_t)255; return r; };
  float* XL   = (float*)alloc((size_t)NP*128*4);
  float* XR   = (float*)alloc((size_t)NP*128*4);
  float* XRES = (float*)alloc((size_t)NP*128*4);
  float* AGG  = (float*)alloc((size_t)NP*128*4);
  float* PCUR = (float*)alloc((size_t)NP*128*4);
  float* PGAT = (float*)alloc((size_t)NP*128*4);
  float* XPF  = (float*)alloc((size_t)NP*128*4);
  float* XP2  = (float*)alloc((size_t)NP*128*4);
  float* A1   = (float*)alloc((size_t)NP*128*4);
  float* DCUR = (float*)alloc((size_t)ND*128*4);
  float* DGAT = (float*)alloc((size_t)ND*128*4);
  float* A2   = (float*)alloc((size_t)ND*128*4);
  float* LG   = (float*)alloc((size_t)(EP+NP)*4*4);
  float* EXB  = (float*)alloc((size_t)(EP+NP)*4*4);
  float* SM   = (float*)alloc((size_t)NP*4*4);
  float* SD   = (float*)alloc((size_t)NP*4*4);
  float* MU   = (float*)alloc(512*4);
  float* VAR  = (float*)alloc(512*4);
  float* EMEAN= (float*)alloc(64*4);
  float* POOLG= (float*)alloc((size_t)NP*4);
  float* POOLE= (float*)alloc((size_t)NP*4);
  float* PM   = (float*)alloc(B*4);
  float* PDEN = (float*)alloc(B*4);
  float* XPp  = (float*)alloc((size_t)B*128*4);
  float* XDp  = (float*)alloc((size_t)B*128*4);
  float* H1   = (float*)alloc((size_t)B*128*4);
  float* H2   = (float*)alloc((size_t)B*64*4);
  float* H3   = (float*)alloc((size_t)B*32*4);
  int* BSTd   = (int*)alloc(B*4);
  int* BCNd   = (int*)alloc(B*4);
  int* BSTp   = (int*)alloc(B*4);
  int* BCNp   = (int*)alloc(B*4);
  _Float16* ACT16 = (_Float16*)alloc((size_t)NP*512*2);
  _Float16* W16   = (_Float16*)alloc((size_t)128*512*2);
  _Float16* Q16   = (_Float16*)alloc((size_t)NP*128*2);
  _Float16* K16   = (_Float16*)alloc((size_t)NP*128*2);
  _Float16* VT16  = (_Float16*)alloc((size_t)128*(NP+32)*2);

  auto cdiv = [](long a, long b){ return (int)((a + b - 1) / b); };
  auto fillf = [&](float* p, long n, float v){ k_fillf<<<cdiv(n,256),256,0,stream>>>(p, v, (int)n); };

  // launch helper: picks TN specialization (no in-kernel N guards)
  auto gemm = [&](const _Float16* A16, const _Float16* Bt16, const float* b,
                  float* Y, int M, int N, int Kp, int act){
    int ntiles = N >> 4;
    int mtiles = M >> 5;
    if ((ntiles & 3) == 0){
      k_gemm<4><<<cdiv(mtiles,4)*(ntiles/4), 128, 0, stream>>>(A16, Bt16, b, Y, M, N, Kp, act);
    } else {
      k_gemm<2><<<cdiv(mtiles,4)*(ntiles/2), 128, 0, stream>>>(A16, Bt16, b, Y, M, N, Kp, act);
    }
  };
  // linear: Y[M,N] = X[M,K] @ W[K,N] + b (opt), with optional leaky-relu
  auto linear = [&](const float* X, int M, int K, const float* W, const float* b,
                    float* Y, int N, int act){
    int Kp = (K + 31) & ~31;
    k_cvtA<<<cdiv((long)M*Kp,256),256,0,stream>>>(X, K, ACT16, Kp, M*Kp);
    k_cvtT<<<cdiv((long)N*Kp,256),256,0,stream>>>(W, K, N, W16, Kp);
    gemm(ACT16, W16, b, Y, M, N, Kp, act);
  };
  auto bn = [&](float* X, int n, int F, const float* g, const float* b, int act){
    k_bn_stats<<<F,256,0,stream>>>(X, n, F, MU, VAR);
    k_bn_apply<<<cdiv((long)n*F,256),256,0,stream>>>(X, F, MU, VAR, g, b, act, n*F);
  };

  // ---- GATv2 encoder ----
  auto gat_encoder = [&](const float* X0, int Nn, int din0, int nlayers, int pb,
                         const int* ei, int E, const float* eattr, int edim,
                         float* XCUR, float* XOUT){
    k_colmean<<<edim,256,0,stream>>>(eattr, E, edim, EMEAN);
    const float* xin = X0; int din = din0;
    int Etot = E + Nn;
    for (int L = 0; L < nlayers; ++L){
      int p = pb + L*8;
      linear(xin, Nn, din, PF(p+0), PF(p+1), XL,   128, 0);   // Wl, bl
      linear(xin, Nn, din, PF(p+2), PF(p+3), XR,   128, 0);   // Wr, br
      linear(xin, Nn, din, PF(p+7), nullptr, XRES, 128, 0);   // Wres
      fillf(SM, (long)Nn*4, -3.0e38f);
      fillf(SD, (long)Nn*4, 0.f);
      fillf(AGG, (long)Nn*128, 0.f);
      k_gat_logit<<<cdiv((long)Etot*4,256),256,0,stream>>>(XL, XR, ei, E, Nn, eattr, edim,
                                                           EMEAN, PF(p+5), PF(p+4), LG, SM);
      k_gat_exp<<<cdiv((long)Etot*4,256),256,0,stream>>>(LG, ei, E, Nn, SM, EXB, SD);
      k_gat_agg<<<cdiv((long)Etot*128,256),256,0,stream>>>(EXB, SD, XL, ei, E, Nn, AGG);
      k_add3<<<cdiv((long)Nn*128,256),256,0,stream>>>(AGG, PF(p+6), XRES, XCUR, Nn*128);
      int bi = pb + nlayers*8 + L*2;
      bn(XCUR, Nn, 128, PF(bi), PF(bi+1), 1);
      xin = XCUR; din = 128;
    }
    int fin = pb + nlayers*8 + nlayers*2;
    linear(XCUR, Nn, 128, PF(fin), PF(fin+1), XOUT, 128, 0);
  };

  // ---- batch segment tables ----
  k_filli<<<1,64,0,stream>>>(BSTd, 0, B); k_filli<<<1,64,0,stream>>>(BCNd, 0, B);
  k_filli<<<1,64,0,stream>>>(BSTp, 0, B); k_filli<<<1,64,0,stream>>>(BCNp, 0, B);
  k_bseg<<<cdiv(ND,256),256,0,stream>>>(DB, ND, BSTd, BCNd);
  k_bseg<<<cdiv(NP,256),256,0,stream>>>(PB, NP, BSTp, BCNp);

  // ---- encoders ----
  gat_encoder(PF(118), ND, 78, 2, 0,  dEI, ED, PF(119), 6,  DCUR, DGAT);  // x_d
  gat_encoder(PF(120), NP, 41, 4, 22, pEI, EP, PF(121), 32, PCUR, PGAT);  // x_p_gat

  // ---- fusion ----
  linear(PF(122), NP, 128, PF(64), PF(65), XL,   128, 0);     // gvp @ m1v
  linear(XL,      NP, 128, PF(66), PF(67), XR,   128, 0);     // a1 = .. @ m1o
  linear(PGAT,    NP, 128, PF(68), PF(69), XL,   128, 0);     // gat @ m2v
  linear(XL,      NP, 128, PF(70), PF(71), XRES, 128, 0);     // a2 = .. @ m2o
  k_cvtOff<<<cdiv((long)NP*128,256),256,0,stream>>>(PGAT,   128, ACT16, 512, 0,   NP*128);
  k_cvtOff<<<cdiv((long)NP*128,256),256,0,stream>>>(PF(122),128, ACT16, 512, 128, NP*128);
  k_cvtOff<<<cdiv((long)NP*128,256),256,0,stream>>>(XR,     128, ACT16, 512, 256, NP*128);
  k_cvtOff<<<cdiv((long)NP*128,256),256,0,stream>>>(XRES,   128, ACT16, 512, 384, NP*128);
  k_cvtT<<<cdiv((long)128*512,256),256,0,stream>>>(PF(72), 512, 128, W16, 512);
  gemm(ACT16, W16, PF(73), XPF, NP, 128, 512, 0);
  bn(XPF, NP, 128, PF(74), PF(75), 1);
  linear(XPF, NP, 128, PF(76), PF(77), XP2, 128, 0);          // x_p

  // ---- cross attention d2p: queries = x_p, keys/values = x_d ----
  linear(XP2,  NP, 128, PF(78), PF(79), XL,   128, 0);
  linear(DGAT, ND, 128, PF(80), PF(81), XR,   128, 0);
  linear(DGAT, ND, 128, PF(82), PF(83), XRES, 128, 0);
  k_cvtA<<<cdiv((long)NP*128,256),256,0,stream>>>(XL, 128, Q16, 128, NP*128);
  k_cvtA<<<cdiv((long)ND*128,256),256,0,stream>>>(XR, 128, K16, 128, ND*128);
  k_cvtT<<<cdiv((long)128*(ND+32),256),256,0,stream>>>(XRES, ND, 128, VT16, ND+32);
  k_cross_attn<<<NP/16, 32, 0, stream>>>(Q16, K16, VT16, PB, DB, BSTd, BCNd, NP, ND, ND+32, AGG);
  linear(AGG, NP, 128, PF(84), PF(85), A1, 128, 0);

  // ---- cross attention p2d: queries = x_d, keys/values = x_p ----
  linear(DGAT, ND, 128, PF(86), PF(87), XL,   128, 0);
  linear(XP2,  NP, 128, PF(88), PF(89), XR,   128, 0);
  linear(XP2,  NP, 128, PF(90), PF(91), XRES, 128, 0);
  k_cvtA<<<cdiv((long)ND*128,256),256,0,stream>>>(XL, 128, Q16, 128, ND*128);
  k_cvtA<<<cdiv((long)NP*128,256),256,0,stream>>>(XR, 128, K16, 128, NP*128);
  k_cvtT<<<cdiv((long)128*(NP+32),256),256,0,stream>>>(XRES, NP, 128, VT16, NP+32);
  k_cross_attn<<<ND/16, 32, 0, stream>>>(Q16, K16, VT16, DB, PB, BSTp, BCNp, ND, NP, NP+32, AGG);
  linear(AGG, ND, 128, PF(92), PF(93), A2, 128, 0);

  // ---- feature-weight mixing ----
  k_fwmix<<<cdiv((long)NP*128,256),256,0,stream>>>(XP2,  A1, PF(94), NP*128);
  k_fwmix<<<cdiv((long)ND*128,256),256,0,stream>>>(DGAT, A2, PF(95), ND*128);

  // ---- attention pooling ----
  auto pool = [&](const float* X, int Nn, const int* batch, int pb, float* OUTB){
    linear(X, Nn, 128, PF(pb), PF(pb+1), XL, 64, 1);
    k_dotw<<<cdiv(Nn,256),256,0,stream>>>(XL, Nn, 64, PF(pb+2), PF(pb+3), POOLG);
    fillf(PM, B, -3.0e38f); fillf(PDEN, B, 0.f); fillf(OUTB, (long)B*128, 0.f);
    k_pool_max<<<cdiv(Nn,256),256,0,stream>>>(POOLG, batch, Nn, PM);
    k_pool_exp<<<cdiv(Nn,256),256,0,stream>>>(POOLG, batch, Nn, PM, POOLE, PDEN);
    k_pool_agg<<<cdiv((long)Nn*128,256),256,0,stream>>>(POOLE, PDEN, batch, X, Nn, OUTB);
  };
  pool(XP2,  NP, PB, 96,  XPp);
  pool(DGAT, ND, DB, 100, XDp);

  // ---- head ----
  k_cvtOff<<<cdiv((long)B*128,256),256,0,stream>>>(XPp, 128, ACT16, 256, 0,   B*128);
  k_cvtOff<<<cdiv((long)B*128,256),256,0,stream>>>(XDp, 128, ACT16, 256, 128, B*128);
  k_cvtT<<<cdiv((long)128*256,256),256,0,stream>>>(PF(104), 256, 128, W16, 256);
  gemm(ACT16, W16, PF(105), H1, B, 128, 256, 0);
  bn(H1, B, 128, PF(106), PF(107), 1);
  linear(H1, B, 128, PF(108), PF(109), H2, 64, 0);
  bn(H2, B, 64, PF(110), PF(111), 1);
  linear(H2, B, 64, PF(112), PF(113), H3, 32, 0);
  bn(H3, B, 32, PF(114), PF(115), 1);
  k_dotw<<<1,64,0,stream>>>(H3, B, 32, PF(116), PF(117), (float*)d_out);
}
